// SelfAttentionLayer_38139309588950
// MI455X (gfx1250) — compile-verified, matrix operand order checked by reading, not run on hardware
//
#include <hip/hip_runtime.h>
#include <hip/hip_bf16.h>

typedef __bf16 bf16_t;
typedef bf16_t v16bf __attribute__((ext_vector_type(16)));
typedef float  v8f   __attribute__((ext_vector_type(8)));

#define S_LEN 2048
#define EMB   1024
#define NHEAD 16
#define HDIM  64
#define LRPE  512

union FragU {
  v16bf v;
  uint4 q[2];
  unsigned short s[16];
};

__device__ __forceinline__ unsigned short f2bf(float x) {
  unsigned u = __builtin_bit_cast(unsigned, x);
  u += 0x7fffu + ((u >> 16) & 1u);
  return (unsigned short)(u >> 16);
}

__device__ __forceinline__ v8f zero8() {
  v8f z;
#pragma unroll
  for (int i = 0; i < 8; ++i) z[i] = 0.0f;
  return z;
}

// Load a 16x32 bf16 fragment (A or B convention) for this lane from a
// row-major matrix: caller passes pointer to (row_of_lane, k_base_of_lane).
// Elements 0..7 = k..k+7, elements 8..15 = k+16..k+23 (ISA 16-bit layout).
__device__ __forceinline__ v16bf ld_frag(const unsigned short* p) {
  FragU u;
  const uint4* q = reinterpret_cast<const uint4*>(p);
  u.q[0] = q[0];   // elements +0  .. +7
  u.q[1] = q[2];   // elements +16 .. +23
  return u.v;
}

__device__ __forceinline__ v8f wmma_bf16(v16bf a, v16bf b, v8f c) {
  return __builtin_amdgcn_wmma_f32_16x16x32_bf16(false, a, false, b,
                                                 (short)0, c, false, false);
}

// ---------------------------------------------------------------- convert
__global__ void cvt_bf16_kernel(const float* __restrict__ src,
                                unsigned short* __restrict__ dst, int n) {
  int i = blockIdx.x * blockDim.x + threadIdx.x;
  if (i < n) dst[i] = f2bf(src[i]);
}

// ------------------------------------------------- projection GEMM (bf16 out)
// out[m][n] = f2bf( (sum_k A[m][k]*W[n][k] + bias[n]) * scale )
// M=2048, N=K=1024. One wave per 64x32 tile (8 C-frags = 64 VGPRs, no spill).
__global__ __launch_bounds__(32) void gemm_proj_kernel(
    const unsigned short* __restrict__ A, const unsigned short* __restrict__ W,
    const float* __restrict__ bias, float scale,
    unsigned short* __restrict__ out) {
  const int lane = threadIdx.x;
  const int lm = lane & 15;
  const int lk = (lane >> 4) * 8;
  const int mh = (lane >> 4) * 8;
  const int m0 = blockIdx.x * 64;
  const int n0 = blockIdx.y * 32;

  const unsigned short* ap[4];
  const unsigned short* bp[2];
#pragma unroll
  for (int t = 0; t < 4; ++t) ap[t] = A + (size_t)(m0 + t * 16 + lm) * EMB + lk;
#pragma unroll
  for (int t = 0; t < 2; ++t) bp[t] = W + (size_t)(n0 + t * 16 + lm) * EMB + lk;

  v8f c[4][2];
#pragma unroll
  for (int i = 0; i < 4; ++i)
#pragma unroll
    for (int j = 0; j < 2; ++j) c[i][j] = zero8();

  for (int kc = 0; kc < EMB; kc += 32) {
    v16bf a[4], b[2];
#pragma unroll
    for (int t = 0; t < 4; ++t) a[t] = ld_frag(ap[t] + kc);
#pragma unroll
    for (int t = 0; t < 2; ++t) b[t] = ld_frag(bp[t] + kc);
#pragma unroll
    for (int mt = 0; mt < 4; ++mt)
#pragma unroll
      for (int nt = 0; nt < 2; ++nt)
        c[mt][nt] = wmma_bf16(a[mt], b[nt], c[mt][nt]);
  }

#pragma unroll
  for (int mt = 0; mt < 4; ++mt)
#pragma unroll
    for (int nt = 0; nt < 2; ++nt) {
      int n = n0 + nt * 16 + lm;
      float bv = bias[n];
#pragma unroll
      for (int i = 0; i < 8; ++i) {
        int m = m0 + mt * 16 + mh + i;
        out[(size_t)m * EMB + n] = f2bf((c[mt][nt][i] + bv) * scale);
      }
    }
}

// ------------------------------------------------- attention (flash-style)
// One wave per (16-query tile, head). qrpe gathered from LDS.
__global__ __launch_bounds__(32) void attn_kernel(
    const unsigned short* __restrict__ qB, const unsigned short* __restrict__ kB,
    const unsigned short* __restrict__ vB,
    const unsigned short* __restrict__ rpeB, const int* __restrict__ pos,
    const float* __restrict__ mask, unsigned short* __restrict__ oB) {
  __shared__ __align__(16) float qrpe_s[16][LRPE];           // 32 KB
  __shared__ __align__(16) unsigned short vtile[32][HDIM];   // 4 KB
  __shared__ __align__(16) unsigned short ptile[16][32];     // 1 KB

  const int lane = threadIdx.x;
  const int lm = lane & 15;
  const int hk = lane >> 4;
  const int lk = hk * 8;
  const int mh = hk * 8;
  const int q0 = blockIdx.x * 16;
  const int h = blockIdx.y;

  // Q fragments: 16 queries x 64 dims as two 16x32 bf16 chunks (pre-scaled).
  v16bf aq[2];
#pragma unroll
  for (int dc = 0; dc < 2; ++dc)
    aq[dc] = ld_frag(qB + (size_t)(q0 + lm) * EMB + h * HDIM + dc * 32 + lk);

  // qrpe[q, l] = q . rpe[l] for this head -> LDS (64 WMMAs)
  for (int lt = 0; lt < LRPE / 16; ++lt) {
    v8f cr = zero8();
#pragma unroll
    for (int dc = 0; dc < 2; ++dc) {
      v16bf br =
          ld_frag(rpeB + (size_t)(lt * 16 + lm) * EMB + h * HDIM + dc * 32 + lk);
      cr = wmma_bf16(aq[dc], br, cr);
    }
#pragma unroll
    for (int i = 0; i < 8; ++i) qrpe_s[mh + i][lt * 16 + lm] = cr[i];
  }
  __syncthreads();

  float mrow[8], lrow[8];
#pragma unroll
  for (int i = 0; i < 8; ++i) { mrow[i] = -1e30f; lrow[i] = 0.0f; }
  v8f oacc[4];
#pragma unroll
  for (int dt = 0; dt < 4; ++dt) oacc[dt] = zero8();

  for (int k0 = 0; k0 < S_LEN; k0 += 32) {
    // Stage V tile (32 keys x 64 dims) into LDS, one row per lane, coalesced.
    {
      const uint4* src =
          reinterpret_cast<const uint4*>(vB + (size_t)(k0 + lane) * EMB + h * HDIM);
      uint4* dst = reinterpret_cast<uint4*>(&vtile[lane][0]);
#pragma unroll
      for (int j = 0; j < 8; ++j) dst[j] = src[j];
    }

    // Score fragments: init from rpe gather (LDS) + mask, then Q*K^T.
    v8f sf[2];
#pragma unroll
    for (int nt = 0; nt < 2; ++nt) {
#pragma unroll
      for (int i = 0; i < 8; ++i) {
        int qq = q0 + mh + i;
        int kk = k0 + nt * 16 + lm;
        int p = pos[(size_t)qq * S_LEN + kk];
        float mv = mask[(size_t)qq * S_LEN + kk];
        sf[nt][i] = qrpe_s[mh + i][p] - (1.0f - mv) * 1e6f;
      }
    }
#pragma unroll
    for (int nt = 0; nt < 2; ++nt) {
#pragma unroll
      for (int dc = 0; dc < 2; ++dc) {
        v16bf bk =
            ld_frag(kB + (size_t)(k0 + nt * 16 + lm) * EMB + h * HDIM + dc * 32 + lk);
        sf[nt] = wmma_bf16(aq[dc], bk, sf[nt]);
      }
    }

    // Online softmax over this 32-key chunk.
#pragma unroll
    for (int i = 0; i < 8; ++i) {
      float mx = fmaxf(sf[0][i], sf[1][i]);
#pragma unroll
      for (int d = 1; d < 16; d <<= 1) mx = fmaxf(mx, __shfl_xor(mx, d, 32));
      float mn = fmaxf(mrow[i], mx);
      float scl = __expf(mrow[i] - mn);
      mrow[i] = mn;
      lrow[i] *= scl;
#pragma unroll
      for (int dt = 0; dt < 4; ++dt) oacc[dt][i] *= scl;
      float p0 = __expf(sf[0][i] - mn);
      float p1 = __expf(sf[1][i] - mn);
      float ps = p0 + p1;
#pragma unroll
      for (int d = 1; d < 16; d <<= 1) ps += __shfl_xor(ps, d, 32);
      lrow[i] += ps;
      ptile[mh + i][lm] = f2bf(p0);
      ptile[mh + i][16 + lm] = f2bf(p1);
    }
    __syncthreads();

    // P (C-layout -> A-layout) via LDS, then P*V.
    FragU pu;
    pu.q[0] = *reinterpret_cast<const uint4*>(&ptile[lm][lk]);
    pu.q[1] = *reinterpret_cast<const uint4*>(&ptile[lm][16 + lk]);
#pragma unroll
    for (int dt = 0; dt < 4; ++dt) {
      FragU vbf;
#pragma unroll
      for (int j = 0; j < 8; ++j) {
        vbf.s[j] = vtile[lk + j][dt * 16 + lm];
        vbf.s[8 + j] = vtile[16 + lk + j][dt * 16 + lm];
      }
      oacc[dt] = wmma_bf16(pu.v, vbf.v, oacc[dt]);
    }
    __syncthreads();
  }

  // Normalize and store bf16 output [s][h*64+d].
#pragma unroll
  for (int dt = 0; dt < 4; ++dt)
#pragma unroll
    for (int i = 0; i < 8; ++i) {
      float val = oacc[dt][i] / lrow[i];
      oB[(size_t)(q0 + mh + i) * EMB + h * HDIM + dt * 16 + lm] = f2bf(val);
    }
}

// ---------------------------------- output GEMM: oB@Wo^T + query@Wqt^T + biases
__global__ __launch_bounds__(32) void gemm_final_kernel(
    const unsigned short* __restrict__ A1, const unsigned short* __restrict__ W1,
    const float* __restrict__ b1, const unsigned short* __restrict__ A2,
    const unsigned short* __restrict__ W2, const float* __restrict__ b2,
    float* __restrict__ out) {
  const int lane = threadIdx.x;
  const int lm = lane & 15;
  const int lk = (lane >> 4) * 8;
  const int mh = (lane >> 4) * 8;
  const int m0 = blockIdx.x * 64;
  const int n0 = blockIdx.y * 32;

  v8f c[4][2];
#pragma unroll
  for (int i = 0; i < 4; ++i)
#pragma unroll
    for (int j = 0; j < 2; ++j) c[i][j] = zero8();

  for (int pp = 0; pp < 2; ++pp) {
    const unsigned short* A = pp ? A2 : A1;
    const unsigned short* W = pp ? W2 : W1;
    const unsigned short* ap[4];
    const unsigned short* bp[2];
#pragma unroll
    for (int t = 0; t < 4; ++t)
      ap[t] = A + (size_t)(m0 + t * 16 + lm) * EMB + lk;
#pragma unroll
    for (int t = 0; t < 2; ++t)
      bp[t] = W + (size_t)(n0 + t * 16 + lm) * EMB + lk;
    for (int kc = 0; kc < EMB; kc += 32) {
      v16bf a[4], b[2];
#pragma unroll
      for (int t = 0; t < 4; ++t) a[t] = ld_frag(ap[t] + kc);
#pragma unroll
      for (int t = 0; t < 2; ++t) b[t] = ld_frag(bp[t] + kc);
#pragma unroll
      for (int mt = 0; mt < 4; ++mt)
#pragma unroll
        for (int nt = 0; nt < 2; ++nt)
          c[mt][nt] = wmma_bf16(a[mt], b[nt], c[mt][nt]);
    }
  }

#pragma unroll
  for (int mt = 0; mt < 4; ++mt)
#pragma unroll
    for (int nt = 0; nt < 2; ++nt) {
      int n = n0 + nt * 16 + lm;
      float bv = b1[n] + b2[n];
#pragma unroll
      for (int i = 0; i < 8; ++i) {
        int m = m0 + mt * 16 + mh + i;
        out[(size_t)m * EMB + n] = c[mt][nt][i] + bv;
      }
    }
}

// ------------------------------------------------------------- layernorm
__global__ __launch_bounds__(256) void ln_kernel(const float* __restrict__ pre,
                                                 const float* __restrict__ gamma,
                                                 const float* __restrict__ beta,
                                                 float* __restrict__ out) {
  int row = blockIdx.x;
  int t = threadIdx.x;
  __shared__ float s1[256];
  __shared__ float s2[256];
  float a = 0.0f, b = 0.0f;
  for (int j = t; j < EMB; j += 256) {
    float x = pre[(size_t)row * EMB + j];
    a += x;
    b += x * x;
  }
  s1[t] = a;
  s2[t] = b;
  __syncthreads();
  for (int st = 128; st > 0; st >>= 1) {
    if (t < st) { s1[t] += s1[t + st]; s2[t] += s2[t + st]; }
    __syncthreads();
  }
  float mu = s1[0] * (1.0f / EMB);
  float var = s2[0] * (1.0f / EMB) - mu * mu;
  float r = rsqrtf(var + 1e-6f);
  for (int j = t; j < EMB; j += 256) {
    float x = pre[(size_t)row * EMB + j];
    out[(size_t)row * EMB + j] = (x - mu) * r * gamma[j] + beta[j];
  }
}

// ------------------------------------------------------------- launcher
extern "C" void kernel_launch(void* const* d_in, const int* in_sizes, int n_in,
                              void* d_out, int out_size, void* d_ws,
                              size_t ws_size, hipStream_t stream) {
  (void)in_sizes; (void)n_in; (void)out_size; (void)ws_size;
  const float* query = (const float*)d_in[0];
  const float* key_  = (const float*)d_in[1];
  const float* value = (const float*)d_in[2];
  const int*   pos   = (const int*)d_in[3];
  const float* mask  = (const float*)d_in[4];
  const float* Wq  = (const float*)d_in[5];
  const float* bq  = (const float*)d_in[6];
  const float* Wk  = (const float*)d_in[7];
  const float* bk  = (const float*)d_in[8];
  const float* Wv  = (const float*)d_in[9];
  const float* bv  = (const float*)d_in[10];
  const float* rpe = (const float*)d_in[11];
  const float* Wqt = (const float*)d_in[12];
  const float* bqt = (const float*)d_in[13];
  const float* Wo  = (const float*)d_in[14];
  const float* bo  = (const float*)d_in[15];
  const float* gamma = (const float*)d_in[16];
  const float* beta  = (const float*)d_in[17];
  float* out = (float*)d_out;

  char* ws = (char*)d_ws;
  size_t off = 0;
  auto alloc = [&](size_t bytes) -> void* {
    void* p = (void*)(ws + off);
    off += (bytes + 255) & ~(size_t)255;
    return p;
  };
  const size_t NE = (size_t)S_LEN * EMB;  // 2M elems
  const size_t NW = (size_t)EMB * EMB;    // 1M elems
  const size_t NR = (size_t)LRPE * EMB;   // 512K elems

  unsigned short* queryB = (unsigned short*)alloc(NE * 2);
  unsigned short* keyB   = (unsigned short*)alloc(NE * 2);
  unsigned short* valueB = (unsigned short*)alloc(NE * 2);
  unsigned short* qB = (unsigned short*)alloc(NE * 2);
  unsigned short* kB = (unsigned short*)alloc(NE * 2);
  unsigned short* vB = (unsigned short*)alloc(NE * 2);
  unsigned short* oB = (unsigned short*)alloc(NE * 2);
  unsigned short* WqB  = (unsigned short*)alloc(NW * 2);
  unsigned short* WkB  = (unsigned short*)alloc(NW * 2);
  unsigned short* WvB  = (unsigned short*)alloc(NW * 2);
  unsigned short* WoB  = (unsigned short*)alloc(NW * 2);
  unsigned short* WqtB = (unsigned short*)alloc(NW * 2);
  unsigned short* rpeB = (unsigned short*)alloc(NR * 2);
  float* pre = (float*)alloc(NE * 4);

  auto cvt = [&](const float* s, unsigned short* d, size_t n) {
    cvt_bf16_kernel<<<dim3((unsigned)((n + 255) / 256)), dim3(256), 0, stream>>>(
        s, d, (int)n);
  };
  cvt(query, queryB, NE);
  cvt(key_, keyB, NE);
  cvt(value, valueB, NE);
  cvt(Wq, WqB, NW);
  cvt(Wk, WkB, NW);
  cvt(Wv, WvB, NW);
  cvt(Wo, WoB, NW);
  cvt(Wqt, WqtB, NW);
  cvt(rpe, rpeB, NR);

  dim3 gg(S_LEN / 64, EMB / 32);
  gemm_proj_kernel<<<gg, 32, 0, stream>>>(queryB, WqB, bq, 0.125f, qB);  // d^-0.5
  gemm_proj_kernel<<<gg, 32, 0, stream>>>(keyB, WkB, bk, 1.0f, kB);
  gemm_proj_kernel<<<gg, 32, 0, stream>>>(valueB, WvB, bv, 1.0f, vB);

  attn_kernel<<<dim3(S_LEN / 16, NHEAD), 32, 0, stream>>>(qB, kB, vB, rpeB, pos,
                                                          mask, oB);

  gemm_final_kernel<<<gg, 32, 0, stream>>>(oB, WoB, bo, queryB, WqtB, bqt, pre);

  ln_kernel<<<dim3(S_LEN), 256, 0, stream>>>(pre, gamma, beta, out);
}
